// GroupedQueryAttention_8186207666325
// MI455X (gfx1250) — matrix operand order repaired and checked by reading, not compiled, not run on hardware
//
#include <hip/hip_runtime.h>
#include <hip/hip_bf16.h>

// ---------------------------------------------------------------------------
// GQA (B=2,S=2048,EMB=2048,H=32,Hkv=8,D=64) for MI455X / gfx1250 (wave32).
// bf16 WMMA (V_WMMA_F32_16X16X32_BF16) for all matmuls, TDM tensor_load_to_lds
// for K tiles in the attention kernel, LDS-staged V^T, online softmax.
// ---------------------------------------------------------------------------

#define S_LEN   2048
#define EMB_    2048
#define NHEADS  32
#define NKV     8
#define HD      64
#define BSZ     2

typedef __bf16        v16bf __attribute__((ext_vector_type(16)));
typedef float         v8f   __attribute__((ext_vector_type(8)));
typedef unsigned int  v4u   __attribute__((ext_vector_type(4)));
typedef int           v8i   __attribute__((ext_vector_type(8)));
typedef int           v4i   __attribute__((ext_vector_type(4)));

#if __has_builtin(__builtin_amdgcn_tensor_load_to_lds)
#define HAVE_TDM 1
#else
#define HAVE_TDM 0
#endif

// float -> bf16 round-to-nearest-even (bit-level)
static __device__ __forceinline__ unsigned short f2bf(float f) {
  unsigned int u = __float_as_uint(f);
  u = (u + 0x7FFFu + ((u >> 16) & 1u)) >> 16;
  return (unsigned short)u;
}

static __device__ __forceinline__ uint4 ld4(const unsigned short* p) {
  return *reinterpret_cast<const uint4*>(p);
}

struct U8x2 { uint4 a, b; };
static __device__ __forceinline__ v16bf make_frag(uint4 x, uint4 y) {
  U8x2 t{x, y};
  return __builtin_bit_cast(v16bf, t);
}

static __device__ __forceinline__ v8f wmma_bf16(v16bf a, v16bf b, v8f c) {
  return __builtin_amdgcn_wmma_f32_16x16x32_bf16(
      /*neg_a=*/false, a, /*neg_b=*/false, b,
      /*c_mod=*/(short)0, c, /*reuse_a=*/false, /*reuse_b=*/false);
}

// ---------------------------------------------------------------------------
// Elementwise f32 -> bf16
// ---------------------------------------------------------------------------
__global__ void cast_bf16_kernel(const float* __restrict__ x,
                                 unsigned short* __restrict__ y, int n) {
  int i = blockIdx.x * blockDim.x + threadIdx.x;
  if (i < n) y[i] = f2bf(x[i]);
}

// W[K,N] f32 -> WT[N,K] bf16
__global__ void transpose_bf16_kernel(const float* __restrict__ W,
                                      unsigned short* __restrict__ WT,
                                      int K, int N) {
  int i = blockIdx.x * blockDim.x + threadIdx.x;
  if (i < K * N) {
    int k = i / N, n = i - k * N;
    WT[(size_t)n * K + k] = f2bf(W[i]);
  }
}

// ---------------------------------------------------------------------------
// GEMM: C[M,N] = A[M,K] * W[K,N], with W given pre-transposed as WT[N,K].
// Workgroup = 4 waves = 64x64 output tile. Wave w owns rows [w*16, w*16+16),
// and 4 C tiles across N. K stepped by 32 (bf16 WMMA K).
// ---------------------------------------------------------------------------
template <bool F32OUT>
__global__ __launch_bounds__(128) void gemm_bf16_kernel(
    const unsigned short* __restrict__ A,   // [M,K] bf16
    const unsigned short* __restrict__ WT,  // [N,K] bf16
    void* __restrict__ Cout, int M, int N, int K) {
  const int nt0  = blockIdx.x * 64;
  const int mt0  = blockIdx.y * 64;
  const int w    = threadIdx.x >> 5;
  const int lane = threadIdx.x & 31;
  const int lg   = lane >> 4;
  const int ln   = lane & 15;
  const int m0   = mt0 + w * 16;

  v8f acc[4];
#pragma unroll
  for (int nt = 0; nt < 4; ++nt) acc[nt] = v8f{0.f,0.f,0.f,0.f,0.f,0.f,0.f,0.f};

  const unsigned short* arow = A + (size_t)(m0 + ln) * K;

  for (int kc = 0; kc < K; kc += 32) {
    // A fragment 16x32: vgprs0-3 -> k = kc + lg*8 .. +7 ; vgprs4-7 -> +16
    v16bf af = make_frag(ld4(arow + kc + lg * 8), ld4(arow + kc + 16 + lg * 8));
#pragma unroll
    for (int nt = 0; nt < 4; ++nt) {
      // B fragment 32x16: lane n = col, k = kc + lg*8 .. +7 ; then +16
      const unsigned short* wrow =
          WT + (size_t)(nt0 + nt * 16 + ln) * K + kc + lg * 8;
      v16bf bf_ = make_frag(ld4(wrow), ld4(wrow + 16));
      acc[nt] = wmma_bf16(af, bf_, acc[nt]);
    }
  }

#pragma unroll
  for (int nt = 0; nt < 4; ++nt) {
#pragma unroll
    for (int v = 0; v < 8; ++v) {
      int m   = v + 8 * lg;              // C/D layout: M = vgpr + 8*lanegroup
      int col = nt0 + nt * 16 + ln;
      if constexpr (F32OUT)
        ((float*)Cout)[(size_t)(m0 + m) * N + col] = acc[nt][v];
      else
        ((unsigned short*)Cout)[(size_t)(m0 + m) * N + col] = f2bf(acc[nt][v]);
    }
  }
}

// ---------------------------------------------------------------------------
// Flash-style GQA attention.
// grid = (S/64 qtiles, H, B), block = 128 (4 waves). Wave w: 16 q rows.
// K tile (64 kv x 64 d) -> LDS via TDM (natural layout, fits score B frag).
// V tile -> LDS transposed [d][kv] so PV B frag is contiguous.
// ---------------------------------------------------------------------------
__global__ __launch_bounds__(128) void gqa_attn_kernel(
    const unsigned short* __restrict__ Qp,  // [B*S, EMB] bf16
    const unsigned short* __restrict__ Kp,  // [B*S, NKV*HD] bf16
    const unsigned short* __restrict__ Vp,  // [B*S, NKV*HD] bf16
    unsigned short* __restrict__ Ctx) {     // [B*S, EMB] bf16
  __shared__ __align__(16) unsigned short kTile[64 * 64];      // [kv][d]
  __shared__ __align__(16) unsigned short vTile[64 * 64];      // [d][kv]
  __shared__ __align__(16) unsigned short pScr[4][16 * 32];    // per-wave P

  const int qt   = blockIdx.x;
  const int h    = blockIdx.y;
  const int b    = blockIdx.z;
  const int kvh  = h >> 2;  // NUM_Q_PER_KV = 4
  const int w    = threadIdx.x >> 5;
  const int lane = threadIdx.x & 31;
  const int lg   = lane >> 4;
  const int ln   = lane & 15;
  const int rowbase = b * S_LEN + qt * 64 + w * 16;

  // Q A-fragments for d-chunks [0,32) and [32,64), held for the whole loop
  const unsigned short* qrow = Qp + (size_t)(rowbase + ln) * EMB_ + h * HD;
  const v16bf aq0 = make_frag(ld4(qrow + lg * 8),      ld4(qrow + 16 + lg * 8));
  const v16bf aq1 = make_frag(ld4(qrow + 32 + lg * 8), ld4(qrow + 48 + lg * 8));

  float mrow[8], lrow[8];
  v8f o[4];
#pragma unroll
  for (int i = 0; i < 8; ++i) { mrow[i] = -1e30f; lrow[i] = 0.f; }
#pragma unroll
  for (int i = 0; i < 4; ++i) o[i] = v8f{0.f,0.f,0.f,0.f,0.f,0.f,0.f,0.f};

  for (int kt = 0; kt < S_LEN / 64; ++kt) {
    const int kv0 = kt * 64;
    __syncthreads();  // previous tile fully consumed

#if HAVE_TDM
    if (w == 0) {
      // Tensor DMA: 64x64 bf16 tile of K (rows kv0..+63, cols kvh*64..+63)
      unsigned long long ga = (unsigned long long)(size_t)(
          Kp + (size_t)(b * S_LEN + kv0) * (NKV * HD) + kvh * HD);
      v4u g0; v8i g1;
      v4i gz4 = {0, 0, 0, 0};
      v8i gz8 = {0, 0, 0, 0, 0, 0, 0, 0};
      g0[0] = 1u;                                   // count=1, user mode
      g0[1] = (unsigned)(size_t)(&kTile[0]);        // lds_addr
      g0[2] = (unsigned)ga;                         // global_addr[31:0]
      g0[3] = (unsigned)((ga >> 32) & 0x1FFFFFFull) // global_addr[56:32]
              | (2u << 30);                         // type=2 ("image")
      g1[0] = (1 << 16);      // data_size=1 (2 bytes)
      g1[1] = (64 << 16);     // tensor_dim0[15:0] = 64
      g1[2] = (64 << 16);     // tensor_dim0 hi=0 | tensor_dim1[15:0]=64
      g1[3] = (64 << 16);     // tensor_dim1 hi=0 | tile_dim0=64
      g1[4] = 64;             // tile_dim1=64 | tile_dim2=0
      g1[5] = NKV * HD;       // tensor_dim0_stride = 512 elements
      g1[6] = 0;
      g1[7] = 0;
      __builtin_amdgcn_tensor_load_to_lds(g0, g1, gz4, gz4, gz8, 0);
    }
#else
    // Fallback: cooperative load of K tile, natural [kv][d] layout
    for (int i = 0; i < 4; ++i) {
      int idx = threadIdx.x + i * 128;
      int r = idx >> 3, c8 = (idx & 7) * 8;
      uint4 qv = ld4(Kp + (size_t)(b * S_LEN + kv0 + r) * (NKV * HD) +
                     kvh * HD + c8);
      *reinterpret_cast<uint4*>(&kTile[r * 64 + c8]) = qv;
    }
#endif

    // Cooperative V load, transposed into LDS as [d][kv]
    for (int i = 0; i < 4; ++i) {
      int idx = threadIdx.x + i * 128;
      int r = idx >> 3, c8 = (idx & 7) * 8;
      uint4 qv = ld4(Vp + (size_t)(b * S_LEN + kv0 + r) * (NKV * HD) +
                     kvh * HD + c8);
      union { uint4 q; unsigned short s[8]; } u;
      u.q = qv;
#pragma unroll
      for (int j = 0; j < 8; ++j) vTile[(c8 + j) * 64 + r] = u.s[j];
    }

#if HAVE_TDM
    if (w == 0) __builtin_amdgcn_s_wait_tensorcnt(0);
#endif
    __syncthreads();

#pragma unroll
    for (int cc = 0; cc < 2; ++cc) {
      const int c = cc * 32;  // kv sub-chunk of 32 within the 64-tile
      v8f s0 = v8f{0.f,0.f,0.f,0.f,0.f,0.f,0.f,0.f};
      v8f s1 = v8f{0.f,0.f,0.f,0.f,0.f,0.f,0.f,0.f};
      {
        // scores tile 0: kv = c .. c+15
        const unsigned short* kr = &kTile[(c + ln) * 64];
        v16bf b0 = make_frag(ld4(kr + lg * 8), ld4(kr + 16 + lg * 8));
        v16bf b1 = make_frag(ld4(kr + 32 + lg * 8), ld4(kr + 48 + lg * 8));
        s0 = wmma_bf16(aq0, b0, s0);
        s0 = wmma_bf16(aq1, b1, s0);
      }
      {
        // scores tile 1: kv = c+16 .. c+31
        const unsigned short* kr = &kTile[(c + 16 + ln) * 64];
        v16bf b0 = make_frag(ld4(kr + lg * 8), ld4(kr + 16 + lg * 8));
        v16bf b1 = make_frag(ld4(kr + 32 + lg * 8), ld4(kr + 48 + lg * 8));
        s1 = wmma_bf16(aq0, b0, s1);
        s1 = wmma_bf16(aq1, b1, s1);
      }

      // Online softmax per output row (row m = v + 8*lg, 16 lanes hold its N)
#pragma unroll
      for (int v = 0; v < 8; ++v) {
        float a  = s0[v] * 0.125f;  // 1/sqrt(64)
        float bb = s1[v] * 0.125f;
        float t = fmaxf(a, bb);
        t = fmaxf(t, __shfl_xor(t, 1, 16));
        t = fmaxf(t, __shfl_xor(t, 2, 16));
        t = fmaxf(t, __shfl_xor(t, 4, 16));
        t = fmaxf(t, __shfl_xor(t, 8, 16));
        float mnew  = fmaxf(mrow[v], t);
        float alpha = __expf(mrow[v] - mnew);
        mrow[v] = mnew;
        float p0 = __expf(a - mnew);
        float p1 = __expf(bb - mnew);
        float rs = p0 + p1;
        rs += __shfl_xor(rs, 1, 16);
        rs += __shfl_xor(rs, 2, 16);
        rs += __shfl_xor(rs, 4, 16);
        rs += __shfl_xor(rs, 8, 16);
        lrow[v] = lrow[v] * alpha + rs;
#pragma unroll
        for (int nt = 0; nt < 4; ++nt) o[nt][v] *= alpha;
        int m = v + 8 * lg;
        pScr[w][m * 32 + ln]      = f2bf(p0);
        pScr[w][m * 32 + 16 + ln] = f2bf(p1);
      }
      // per-wave LDS round trip: DS ops are in-order per wave; make sure
      // the compiler doesn't move the loads above the stores.
      asm volatile("s_wait_dscnt 0x0" ::: "memory");

      // P as A-fragment (16x32, k = kv within chunk)
      const unsigned short* pr = &pScr[w][ln * 32];
      v16bf ap = make_frag(ld4(pr + lg * 8), ld4(pr + 16 + lg * 8));

      // O += P @ V : B frag from vTile[d][kv], contiguous along kv
#pragma unroll
      for (int nt = 0; nt < 4; ++nt) {
        const unsigned short* vr = &vTile[(nt * 16 + ln) * 64 + c];
        v16bf bv = make_frag(ld4(vr + lg * 8), ld4(vr + 16 + lg * 8));
        o[nt] = wmma_bf16(ap, bv, o[nt]);
      }
    }
  }

  // Normalize and write ctx (bf16)
#pragma unroll
  for (int nt = 0; nt < 4; ++nt) {
#pragma unroll
    for (int v = 0; v < 8; ++v) {
      int m = v + 8 * lg;
      float val = o[nt][v] / lrow[v];
      Ctx[(size_t)(rowbase + m) * EMB_ + h * HD + nt * 16 + ln] = f2bf(val);
    }
  }
}

// ---------------------------------------------------------------------------
// Launch
// ---------------------------------------------------------------------------
extern "C" void kernel_launch(void* const* d_in, const int* in_sizes, int n_in,
                              void* d_out, int out_size, void* d_ws,
                              size_t ws_size, hipStream_t stream) {
  const float* q_f  = (const float*)d_in[0];
  const float* k_f  = (const float*)d_in[1];
  const float* v_f  = (const float*)d_in[2];
  const float* Wq_f = (const float*)d_in[3];
  const float* Wk_f = (const float*)d_in[4];
  const float* Wv_f = (const float*)d_in[5];
  const float* Wo_f = (const float*)d_in[6];

  const size_t MB = 1ull << 20;
  char* ws = (char*)d_ws;
  unsigned short* qb  = (unsigned short*)(ws + 0 * MB);    // 16 MB [4096,2048]
  unsigned short* kb  = (unsigned short*)(ws + 16 * MB);   // 16 MB
  unsigned short* vb  = (unsigned short*)(ws + 32 * MB);   // 16 MB
  unsigned short* WqT = (unsigned short*)(ws + 48 * MB);   //  8 MB [2048,2048]
  unsigned short* WkT = (unsigned short*)(ws + 56 * MB);   //  2 MB [512,2048]
  unsigned short* WvT = (unsigned short*)(ws + 58 * MB);   //  2 MB
  unsigned short* WoT = (unsigned short*)(ws + 60 * MB);   //  8 MB
  unsigned short* Qp  = (unsigned short*)(ws + 68 * MB);   // 16 MB [4096,2048]
  unsigned short* Kp  = (unsigned short*)(ws + 84 * MB);   //  4 MB [4096,512]
  unsigned short* Vp  = (unsigned short*)(ws + 88 * MB);   //  4 MB
  unsigned short* ctx = (unsigned short*)(ws + 92 * MB);   // 16 MB [4096,2048]

  const int nQKV = BSZ * S_LEN * EMB_;  // 8388608
  cast_bf16_kernel<<<nQKV / 256, 256, 0, stream>>>(q_f, qb, nQKV);
  cast_bf16_kernel<<<nQKV / 256, 256, 0, stream>>>(k_f, kb, nQKV);
  cast_bf16_kernel<<<nQKV / 256, 256, 0, stream>>>(v_f, vb, nQKV);

  transpose_bf16_kernel<<<(EMB_ * EMB_) / 256, 256, 0, stream>>>(Wq_f, WqT,
                                                                 EMB_, EMB_);
  transpose_bf16_kernel<<<(EMB_ * NKV * HD) / 256, 256, 0, stream>>>(
      Wk_f, WkT, EMB_, NKV * HD);
  transpose_bf16_kernel<<<(EMB_ * NKV * HD) / 256, 256, 0, stream>>>(
      Wv_f, WvT, EMB_, NKV * HD);
  transpose_bf16_kernel<<<(EMB_ * EMB_) / 256, 256, 0, stream>>>(Wo_f, WoT,
                                                                 EMB_, EMB_);

  const int M = BSZ * S_LEN;  // 4096
  // Q/K/V projections
  gemm_bf16_kernel<false><<<dim3(EMB_ / 64, M / 64), 128, 0, stream>>>(
      qb, WqT, Qp, M, EMB_, EMB_);
  gemm_bf16_kernel<false><<<dim3((NKV * HD) / 64, M / 64), 128, 0, stream>>>(
      kb, WkT, Kp, M, NKV * HD, EMB_);
  gemm_bf16_kernel<false><<<dim3((NKV * HD) / 64, M / 64), 128, 0, stream>>>(
      vb, WvT, Vp, M, NKV * HD, EMB_);

  // Attention
  gqa_attn_kernel<<<dim3(S_LEN / 64, NHEADS, BSZ), 128, 0, stream>>>(Qp, Kp,
                                                                     Vp, ctx);

  // Output projection (f32 out)
  gemm_bf16_kernel<true><<<dim3(EMB_ / 64, M / 64), 128, 0, stream>>>(
      ctx, WoT, d_out, M, EMB_, EMB_);
}
